// multi_MLP_43018392437065
// MI455X (gfx1250) — matrix operand order repaired and checked by reading, not compiled, not run on hardware
//
#include <hip/hip_runtime.h>
#include <hip/hip_bf16.h>

// ---------- types ----------
typedef __attribute__((ext_vector_type(16))) __bf16         bf16x16;
typedef __attribute__((ext_vector_type(8)))  float          v8f;
typedef __attribute__((ext_vector_type(8)))  unsigned short u16x8;
typedef __attribute__((ext_vector_type(16))) unsigned short u16x16;

// Problem constants
#define BATCH   256
#define CCH     64
#define HH      32
#define SS      8
#define LL      16
#define DIN     4096
#define DTMP    1024
#define DOUT    4096

// Tiling
#define TM      128
#define TN      128
#define TK      64         // K per staged tile (2 wmma K-steps)
#define LDSPAD  72         // 144B row stride: 16B aligned, conflict-free frag reads

// ---------- bf16 conversion ----------
__device__ __forceinline__ unsigned short f2bf1(float f) {
    unsigned u = __float_as_uint(f);
    u = (u + 0x7FFFu + ((u >> 16) & 1u)) >> 16;
    return (unsigned short)u;
}
// gfx1250 packed f32->bf16 convert (RNE) — single VALU op.
__device__ __forceinline__ unsigned f2bf_pk(float a, float b) {
    unsigned r;
    asm("v_cvt_pk_bf16_f32 %0, %1, %2" : "=v"(r) : "v"(a), "v"(b));
    return r;
}

// ---------- gfx1250 async copy: global -> LDS, 16B per lane ----------
__device__ __forceinline__ void async_copy_b128(void* lds_ptr, const void* gptr) {
    unsigned           lds_off = (unsigned)(unsigned long long)lds_ptr; // addr[31:0] = LDS offset
    unsigned long long ga      = (unsigned long long)gptr;
    asm volatile("global_load_async_to_lds_b128 %0, %1, off"
                 :: "v"(lds_off), "v"(ga) : "memory");
}
__device__ __forceinline__ void async_wait0() {
    asm volatile("s_wait_asynccnt 0x0" ::: "memory");
}

// ---------- WMMA fragment load (two contiguous 16B LDS chunks) ----------
__device__ __forceinline__ bf16x16 load_frag(const unsigned short* p0,
                                             const unsigned short* p1) {
    u16x8 lo = *(const u16x8*)p0;
    u16x8 hi = *(const u16x8*)p1;
    union { u16x16 u; bf16x16 b; } cv;
    cv.u = __builtin_shufflevector(lo, hi, 0,1,2,3,4,5,6,7,8,9,10,11,12,13,14,15);
    return cv.b;
}
__device__ __forceinline__ float gelu_erf(float x) {
    return 0.5f * x * (1.0f + erff(x * 0.7071067811865476f));
}

// ---------- staging helpers ----------
// A: pure async copy of a bf16 row-major tile (row stride = KTOT elements)
template<int KTOT>
__device__ __forceinline__ void stageA_copy(unsigned short (*As)[LDSPAD],
                                            const unsigned short* Abase,
                                            int kb, int tid) {
    #pragma unroll
    for (int i = 0; i < 4; ++i) {
        int q = tid + i * 256;                    // 1024 x 16B chunks (128 rows x 8)
        int row = q >> 3, ch = q & 7;
        async_copy_b128(&As[row][ch * 8],
                        Abase + (size_t)row * KTOT + kb + ch * 8);
    }
}
// B: fp32 weights -> bf16, transposed into Bt[n][k]
template<int NTOT, int KTOT>
__device__ __forceinline__ void stageB_conv(unsigned short (*Bt)[LDSPAD],
                                            const float* Wbase,  // + l*K*N + n0
                                            int kb, int tid) {
    #pragma unroll
    for (int i = 0; i < 4; ++i) {
        int q  = tid + i * 256;                   // 32 k-pairs x 32 n-quads
        int kp = q & 31, nq = q >> 5;
        const float* src = Wbase + (size_t)(kb + kp * 2) * NTOT + nq * 4;
        float4 r0 = *(const float4*)src;
        float4 r1 = *(const float4*)(src + NTOT);
        if (kb + 3 * TK < KTOT)
            __builtin_prefetch(src + (size_t)2 * TK * NTOT, 0, 1);
        *(unsigned*)&Bt[nq * 4 + 0][kp * 2] = f2bf_pk(r0.x, r1.x);
        *(unsigned*)&Bt[nq * 4 + 1][kp * 2] = f2bf_pk(r0.y, r1.y);
        *(unsigned*)&Bt[nq * 4 + 2][kp * 2] = f2bf_pk(r0.z, r1.z);
        *(unsigned*)&Bt[nq * 4 + 3][kp * 2] = f2bf_pk(r0.w, r1.w);
    }
}

// ---------- shared inner compute: 2 K-sub-steps of 8 WMMAs ----------
__device__ __forceinline__ void compute_tile(const unsigned short (*As)[LDSPAD],
                                             const unsigned short (*Bt)[LDSPAD],
                                             int wm, int wn, int lmod, int lhalf,
                                             v8f acc[4][2]) {
    #pragma unroll
    for (int ksub = 0; ksub < 2; ++ksub) {
        const int kc = ksub * 32;
        bf16x16 af[4], bfr[2];
        #pragma unroll
        for (int mi = 0; mi < 4; ++mi) {
            const unsigned short* p = &As[wm + mi * 16 + lmod][kc + lhalf * 8];
            af[mi] = load_frag(p, p + 16);
        }
        #pragma unroll
        for (int ni = 0; ni < 2; ++ni) {
            const unsigned short* p = &Bt[wn + ni * 16 + lmod][kc + lhalf * 8];
            bfr[ni] = load_frag(p, p + 16);
        }
        #pragma unroll
        for (int mi = 0; mi < 4; ++mi)
            #pragma unroll
            for (int ni = 0; ni < 2; ++ni)
                acc[mi][ni] = __builtin_amdgcn_wmma_f32_16x16x32_bf16(
                    false, af[mi], false, bfr[ni], (short)0, acc[mi][ni], false, false);
    }
}

// =====================================================================
// Kernel 0: unfold x -> P bf16 [L][B][DIN]
// =====================================================================
__global__ __launch_bounds__(256)
void unfold_x_bf16(const float* __restrict__ x, unsigned short* __restrict__ P) {
    int gid = blockIdx.x * 256 + threadIdx.x;     // 2,097,152 chunks of 8 floats
    int cs  = gid & 511;                          // c*8 + sh
    int b   = (gid >> 9) & 255;
    int l   = gid >> 17;
    int gh  = l >> 2, gw = l & 3;
    int c   = cs >> 3, sh = cs & 7;
    const float* src = x + (size_t)b * (CCH * HH * HH) + (size_t)c * (HH * HH)
                         + (size_t)(gh * SS + sh) * HH + gw * SS;
    float4 f0 = *(const float4*)src;
    float4 f1 = *(const float4*)(src + 4);
    uint4 o;
    o.x = f2bf_pk(f0.x, f0.y);
    o.y = f2bf_pk(f0.z, f0.w);
    o.z = f2bf_pk(f1.x, f1.y);
    o.w = f2bf_pk(f1.z, f1.w);
    *(uint4*)(P + ((size_t)(l * BATCH + b) * 512 + cs) * 8) = o;
}

// =====================================================================
// Kernel 1: GEMM1 (P[l] 256x4096 @ W1[l] 4096x1024) + b1 + erf-GELU -> H bf16
// grid (8, 2, 16), block 256
// =====================================================================
__global__ __launch_bounds__(256)
void mlp_gemm1_gelu(const unsigned short* __restrict__ P,
                    const float* __restrict__ W1,
                    const float* __restrict__ b1,
                    unsigned short* __restrict__ Hbuf) {
    __shared__ unsigned short As[2][TM][LDSPAD];
    __shared__ unsigned short Bt[2][TN][LDSPAD];

    const int tid  = threadIdx.x;
    const int n0   = blockIdx.x * TN;
    const int m0   = blockIdx.y;
    const int l    = blockIdx.z;
    const int lane = tid & 31, wav = tid >> 5;
    const int lmod = lane & 15, lhalf = lane >> 4;
    const int wm   = (wav & 1) * 64;
    const int wn   = (wav >> 1) * 32;

    const unsigned short* Abase = P + ((size_t)l * BATCH + m0 * TM) * DIN;
    const float*          Wbase = W1 + (size_t)l * DIN * DTMP + n0;

    v8f acc[4][2];
    #pragma unroll
    for (int mi = 0; mi < 4; ++mi)
        #pragma unroll
        for (int ni = 0; ni < 2; ++ni) acc[mi][ni] = (v8f){};

    stageA_copy<DIN>(As[0], Abase, 0, tid);
    stageB_conv<DTMP, DIN>(Bt[0], Wbase, 0, tid);
    async_wait0();
    __syncthreads();

    int cur = 0;
    const int NK = DIN / TK;                      // 64
    #pragma unroll 1
    for (int it = 0; it < NK; ++it) {
        if (it + 1 < NK) {
            stageA_copy<DIN>(As[cur ^ 1], Abase, (it + 1) * TK, tid);
            stageB_conv<DTMP, DIN>(Bt[cur ^ 1], Wbase, (it + 1) * TK, tid);
        }
        compute_tile(As[cur], Bt[cur], wm, wn, lmod, lhalf, acc);
        async_wait0();
        __syncthreads();
        cur ^= 1;
    }

    float bias[2];
    int   ncol[2];
    #pragma unroll
    for (int ni = 0; ni < 2; ++ni) {
        ncol[ni] = n0 + wn + ni * 16 + lmod;
        bias[ni] = b1[(size_t)l * DTMP + ncol[ni]];
    }
    #pragma unroll
    for (int mi = 0; mi < 4; ++mi)
        #pragma unroll
        for (int ni = 0; ni < 2; ++ni)
            #pragma unroll
            for (int v = 0; v < 8; ++v) {
                int   mrow = wm + mi * 16 + v + 8 * lhalf;
                float hv   = gelu_erf(acc[mi][ni][v] + bias[ni]);
                Hbuf[(size_t)l * BATCH * DTMP
                     + (size_t)(m0 * TM + mrow) * DTMP + ncol[ni]] = f2bf1(hv);
            }
}

// =====================================================================
// Kernel 2: GEMM2 (H[l] 256x1024 @ W2[l] 1024x4096) + b2, fused fold
// grid (32, 2, 16), block 256
// =====================================================================
__global__ __launch_bounds__(256)
void mlp_gemm2_fold(const unsigned short* __restrict__ Hbuf,
                    const float* __restrict__ W2,
                    const float* __restrict__ b2,
                    float* __restrict__ out) {
    __shared__ unsigned short As[2][TM][LDSPAD];
    __shared__ unsigned short Bt[2][TN][LDSPAD];

    const int tid  = threadIdx.x;
    const int n0   = blockIdx.x * TN;
    const int m0   = blockIdx.y;
    const int l    = blockIdx.z;
    const int gh   = l >> 2, gw = l & 3;
    const int lane = tid & 31, wav = tid >> 5;
    const int lmod = lane & 15, lhalf = lane >> 4;
    const int wm   = (wav & 1) * 64;
    const int wn   = (wav >> 1) * 32;

    const unsigned short* Abase = Hbuf + ((size_t)l * BATCH + m0 * TM) * DTMP;
    const float*          Wbase = W2 + (size_t)l * DTMP * DOUT + n0;

    v8f acc[4][2];
    #pragma unroll
    for (int mi = 0; mi < 4; ++mi)
        #pragma unroll
        for (int ni = 0; ni < 2; ++ni) acc[mi][ni] = (v8f){};

    stageA_copy<DTMP>(As[0], Abase, 0, tid);
    stageB_conv<DOUT, DTMP>(Bt[0], Wbase, 0, tid);
    async_wait0();
    __syncthreads();

    int cur = 0;
    const int NK = DTMP / TK;                     // 16
    #pragma unroll 1
    for (int it = 0; it < NK; ++it) {
        if (it + 1 < NK) {
            stageA_copy<DTMP>(As[cur ^ 1], Abase, (it + 1) * TK, tid);
            stageB_conv<DOUT, DTMP>(Bt[cur ^ 1], Wbase, (it + 1) * TK, tid);
        }
        compute_tile(As[cur], Bt[cur], wm, wn, lmod, lhalf, acc);
        async_wait0();
        __syncthreads();
        cur ^= 1;
    }

    float bias[2];
    int   colbase[2];
    #pragma unroll
    for (int ni = 0; ni < 2; ++ni) {
        int d = n0 + wn + ni * 16 + lmod;
        bias[ni] = b2[(size_t)l * DOUT + d];
        int c = d >> 6, sh = (d >> 3) & 7, sw = d & 7;
        colbase[ni] = c * (HH * HH) + (gh * SS + sh) * HH + gw * SS + sw;
    }
    #pragma unroll
    for (int mi = 0; mi < 4; ++mi)
        #pragma unroll
        for (int ni = 0; ni < 2; ++ni)
            #pragma unroll
            for (int v = 0; v < 8; ++v) {
                int b = m0 * TM + wm + mi * 16 + v + 8 * lhalf;
                out[(size_t)b * (CCH * HH * HH) + colbase[ni]] = acc[mi][ni][v] + bias[ni];
            }
}

// =====================================================================
extern "C" void kernel_launch(void* const* d_in, const int* in_sizes, int n_in,
                              void* d_out, int out_size, void* d_ws, size_t ws_size,
                              hipStream_t stream) {
    (void)in_sizes; (void)n_in; (void)out_size; (void)ws_size;
    const float* x  = (const float*)d_in[0];
    const float* W1 = (const float*)d_in[1];
    const float* b1 = (const float*)d_in[2];
    const float* W2 = (const float*)d_in[3];
    const float* b2 = (const float*)d_in[4];

    unsigned short* P    = (unsigned short*)d_ws;                    // 32 MB
    unsigned short* Hbuf = P + (size_t)LL * BATCH * DIN;             //  8 MB

    unfold_x_bf16<<<dim3((LL * BATCH * 512) / 256), 256, 0, stream>>>(x, P);
    mlp_gemm1_gelu<<<dim3(DTMP / TN, BATCH / TM, LL), 256, 0, stream>>>(P, W1, b1, Hbuf);
    mlp_gemm2_fold<<<dim3(DOUT / TN, BATCH / TM, LL), 256, 0, stream>>>(Hbuf, W2, b2,
                                                                        (float*)d_out);
}
